// CellMatesTransformer_16329465660113
// MI455X (gfx1250) — matrix operand order repaired and checked by reading, not compiled, not run on hardware
//
#include <hip/hip_runtime.h>
#include <math.h>

// Problem constants (match reference)
#define NL 8
#define Bz 4
#define Lz 512
#define Dz 512
#define Hz 16
#define KHz 32
#define FFz 2048
#define MMz 512
#define NDz 14
#define SCALEF 0.17677669529663687f  // 1/sqrt(32)

typedef _Float16 f16;
typedef __attribute__((ext_vector_type(16))) _Float16 v16h;
typedef __attribute__((ext_vector_type(8)))  float    v8f;
typedef unsigned int u32x4 __attribute__((ext_vector_type(4)));
typedef int          i32x8 __attribute__((ext_vector_type(8)));
typedef int          i32x4 __attribute__((ext_vector_type(4)));

#define WMMA(a, b, c) __builtin_amdgcn_wmma_f32_16x16x32_f16( \
    false, (a), false, (b), (short)0, (c), false, false)

// ---------- Tensor Data Mover: 2-D f32 tile, global -> LDS ----------
// D# layout per CDNA5 ISA (group0 128b, group1 256b).
// tile_d0/tensor_d0 = fast (contiguous) dim, stride0 in elements.
__device__ __forceinline__ void tdm_load_2d_f32(unsigned lds_off, const float* gaddr,
                                                unsigned tile_d0, unsigned tile_d1,
                                                unsigned tensor_d0, unsigned tensor_d1,
                                                unsigned stride0) {
    unsigned long long ga = (unsigned long long)(const void*)gaddr;
    u32x4 g0;
    g0[0] = 1u;                                   // count=1, user descriptor
    g0[1] = lds_off;                              // LDS byte address
    g0[2] = (unsigned)(ga & 0xffffffffu);         // global_addr[31:0]
    g0[3] = (unsigned)((ga >> 32) & 0x01ffffffu)  // global_addr[56:32]
            | (2u << 30);                         // type=2 ("image")
    i32x8 g1;
    g1[0] = (int)(2u << 16);                      // data_size=4B; no multicast
    g1[1] = (int)((tensor_d0 & 0xffffu) << 16);   // [63:48] tensor_dim0 lo16
    g1[2] = (int)((tensor_d0 >> 16) | ((tensor_d1 & 0xffffu) << 16));
    g1[3] = (int)((tensor_d1 >> 16) | (tile_d0 << 16));   // [127:112] tile_dim0
    g1[4] = (int)(tile_d1 & 0xffffu);             // tile_dim1; tile_dim2=0
    g1[5] = (int)stride0;                         // tensor_dim0_stride[31:0]
    g1[6] = 0;                                    // stride0 hi16 | stride1 lo16
    g1[7] = 0;
    i32x4 z4 = {0, 0, 0, 0};
    i32x8 z8 = {0, 0, 0, 0, 0, 0, 0, 0};
    __builtin_amdgcn_tensor_load_to_lds(g0, g1, z4, z4, z8, 0);
}

// ---------- WMMA fragment loaders (contiguous k order, wave32), f32 LDS -> f16 frag ----
// A (16xK=32): lane -> M = lane%16 ; elem j -> K = j + 16g, g=lane/16
__device__ __forceinline__ v16h load_a_lds(const float* a, int lda, int lane) {
    int m = lane & 15, g = lane >> 4;
    v16h r;
#pragma unroll
    for (int j = 0; j < 16; ++j) {
        int k = j + 16 * g;
        r[j] = (f16)a[m * lda + k];
    }
    return r;
}
// B (K=32 x 16): lane -> N = lane%16 ; elem j -> K = j + 16g
__device__ __forceinline__ v16h load_b_lds(const float* b, int ldb, int lane) {
    int n = lane & 15, g = lane >> 4;
    v16h r;
#pragma unroll
    for (int j = 0; j < 16; ++j) r[j] = (f16)b[(j + 16 * g) * ldb + n];
    return r;
}

// ---------- small kernels ----------
__global__ void k_embed(const int* __restrict__ ct, const float* __restrict__ emb,
                        float* __restrict__ h) {
    int i = blockIdx.x * blockDim.x + threadIdx.x;
    if (i >= Bz * Lz * Dz) return;
    int bl = i >> 9, d = i & 511;
    h[i] = emb[ct[bl] * Dz + d];
}

__global__ void k_didx(const float* __restrict__ dist, unsigned char* __restrict__ didx) {
    int i = blockIdx.x * blockDim.x + threadIdx.x;
    if (i >= Bz * Lz * Lz) return;
    float v = dist[i];
    int c = 0;
#pragma unroll
    for (int j = 1; j <= 14; ++j) c += (10.0f * (float)j < v) ? 1 : 0;  // searchsorted left
    didx[i] = (unsigned char)(c > NDz - 1 ? NDz - 1 : c);
}

// qb[b,l,h,d] = sum_k q[b,l,h,k] * Ktab[d,k]
__global__ void k_relproj(const float* __restrict__ qk, const float* __restrict__ Ktab,
                          float* __restrict__ outp) {
    int idx = blockIdx.x * blockDim.x + threadIdx.x;
    if (idx >= Bz * Lz * Hz * NDz) return;
    int d = idx % NDz;
    int rest = idx / NDz;
    const float* src = qk + (size_t)rest * KHz;
    const float* kt = Ktab + d * KHz;
    float s = 0.f;
#pragma unroll
    for (int kk = 0; kk < KHz; ++kk) s += src[kk] * kt[kk];
    outp[idx] = s;
}

// ---------- tiled WMMA GEMM with TDM double-buffered staging ----------
// C[M,N] = act(A[M,K] @ B[K,N] + bias); M,N,K multiples of 64/64/32.
template <int ACT>  // 0 = none, 1 = exact gelu
__global__ __launch_bounds__(256) void k_gemm(const float* __restrict__ A,
                                              const float* __restrict__ Bw,
                                              const float* __restrict__ bias,
                                              float* __restrict__ C,
                                              int M, int N, int K) {
    __shared__ float Af[2][64 * 32];  // A tile, row-major [64][32]
    __shared__ float Bf[2][32 * 64];  // B tile, row-major [32][64]
    int n0 = blockIdx.x * 64, m0 = blockIdx.y * 64;
    int t = threadIdx.x, wave = t >> 5, lane = t & 31;
    // 16 sub-tiles (4x4 of 16x16); each of 8 waves owns 2
    int t0 = 2 * wave, t1 = 2 * wave + 1;
    int tm0 = t0 >> 2, tn0 = t0 & 3, tm1 = t1 >> 2, tn1 = t1 & 3;
    v8f acc0 = {}, acc1 = {};

    unsigned ldsA0 = (unsigned)(size_t)&Af[0][0];
    unsigned ldsA1 = (unsigned)(size_t)&Af[1][0];
    unsigned ldsB0 = (unsigned)(size_t)&Bf[0][0];
    unsigned ldsB1 = (unsigned)(size_t)&Bf[1][0];

    int nch = K >> 5;
    if (wave == 0) {  // prologue: DMA chunk 0 into buffer 0
        tdm_load_2d_f32(ldsA0, A + (size_t)m0 * K, 32, 64, (unsigned)K, (unsigned)M,
                        (unsigned)K);
        tdm_load_2d_f32(ldsB0, Bw + n0, 64, 32, (unsigned)N, (unsigned)K, (unsigned)N);
    }
    int pb = 0;
    for (int ci = 0; ci < nch; ++ci) {
        if (wave == 0) {
            if (ci + 1 < nch) {  // kick DMA for next chunk into the other buffer
                unsigned la = pb ? ldsA0 : ldsA1;
                unsigned lb = pb ? ldsB0 : ldsB1;
                int kc1 = (ci + 1) << 5;
                tdm_load_2d_f32(la, A + (size_t)m0 * K + kc1, 32, 64, (unsigned)K,
                                (unsigned)M, (unsigned)K);
                tdm_load_2d_f32(lb, Bw + (size_t)kc1 * N + n0, 64, 32, (unsigned)N,
                                (unsigned)K, (unsigned)N);
                __builtin_amdgcn_s_wait_tensorcnt(2);  // current chunk landed
            } else {
                __builtin_amdgcn_s_wait_tensorcnt(0);
            }
        }
        __syncthreads();
        const float* Ab = &Af[pb][0];
        const float* Bb = &Bf[pb][0];
        v16h a0 = load_a_lds(Ab + tm0 * 16 * 32, 32, lane);
        v16h b0 = load_b_lds(Bb + tn0 * 16, 64, lane);
        acc0 = WMMA(a0, b0, acc0);
        v16h a1 = load_a_lds(Ab + tm1 * 16 * 32, 32, lane);
        v16h b1 = load_b_lds(Bb + tn1 * 16, 64, lane);
        acc1 = WMMA(a1, b1, acc1);
        __syncthreads();
        pb ^= 1;
    }
    int n = lane & 15, g = lane >> 4;
#pragma unroll
    for (int r = 0; r < 8; ++r) {
        {
            int row = m0 + tm0 * 16 + r + 8 * g, col = n0 + tn0 * 16 + n;
            float v = acc0[r] + bias[col];
            if (ACT == 1) v = 0.5f * v * (1.0f + erff(v * 0.70710678118654752f));
            C[(size_t)row * N + col] = v;
        }
        {
            int row = m0 + tm1 * 16 + r + 8 * g, col = n0 + tn1 * 16 + n;
            float v = acc1[r] + bias[col];
            if (ACT == 1) v = 0.5f * v * (1.0f + erff(v * 0.70710678118654752f));
            C[(size_t)row * N + col] = v;
        }
    }
}

// ---------- fused attention: one wave per (b, h, 16-query tile) ----------
__global__ __launch_bounds__(32) void k_attn(
    const float* __restrict__ q, const float* __restrict__ k, const float* __restrict__ v,
    const float* __restrict__ qb, const float* __restrict__ kb,
    const unsigned char* __restrict__ didx,
    const float* __restrict__ Kkr, const float* __restrict__ Vqk,
    const float* __restrict__ Vqr, const float* __restrict__ Vkr,
    float* __restrict__ z) {
    __shared__ float e_lds[16][Lz];   // 32 KB score rows for this q-tile
    __shared__ float aw[16][NDz];     // per-row bucket-weighted attention sums
    __shared__ float vb[NDz][KHz];    // Vqk+Vqr+Vkr
    __shared__ float ekr[NDz];
    __shared__ float rmx[16], rsum[16];

    int bid = blockIdx.x;
    int qt = bid & 31;             // L/16 = 32 tiles
    int h = (bid >> 5) & 15;
    int b = bid >> 9;
    int lane = threadIdx.x;
    int m = lane & 15, g = lane >> 4;
    int l0 = qt * 16;

    for (int i = lane; i < NDz * KHz; i += 32) vb[i >> 5][i & 31] = Vqk[i] + Vqr[i] + Vkr[i];
    if (lane < NDz) {
        float s = 0.f;
#pragma unroll
        for (int kk = 0; kk < KHz; ++kk) s += Kkr[lane * KHz + kk];
        ekr[lane] = s;
    }
    for (int i = lane; i < 16 * NDz; i += 32) aw[i / NDz][i % NDz] = 0.f;
    __syncthreads();

    // Q fragment (A): reused for all key tiles
    v16h aq;
    {
        const float* qp = q + (((size_t)b * Lz + l0 + m) * Hz + h) * KHz;
#pragma unroll
        for (int j = 0; j < 16; ++j) {
            int kk = j + 16 * g;
            aq[j] = (f16)qp[kk];
        }
    }

    // ---- pass 1: scores + biases -> LDS, track row max ----
    float rmax[8];
#pragma unroll
    for (int r = 0; r < 8; ++r) rmax[r] = -1e30f;
    for (int xt = 0; xt < Lz / 16; ++xt) {
        int x = xt * 16 + m;  // key index for this lane (B-fragment N)
        const float* kp = k + (((size_t)b * Lz + x) * Hz + h) * KHz + 16 * g;
        v16h bk;
#pragma unroll
        for (int j = 0; j < 16; ++j) bk[j] = (f16)kp[j];
        v8f s = {};
        s = WMMA(aq, bk, s);
#pragma unroll
        for (int r = 0; r < 8; ++r) {
            int row = r + 8 * g;
            int gl = l0 + row;
            int gx = xt * 16 + m;
            int d = didx[((size_t)b * Lz + gl) * Lz + gx];
            float e = (s[r] + qb[(((size_t)b * Lz + gl) * Hz + h) * NDz + d]
                            + kb[(((size_t)b * Lz + gx) * Hz + h) * NDz + d]
                            + ekr[d]) * SCALEF;
            e_lds[row][gx] = e;
            rmax[r] = fmaxf(rmax[r], e);
        }
    }
#pragma unroll
    for (int r = 0; r < 8; ++r)
        for (int off = 8; off; off >>= 1)
            rmax[r] = fmaxf(rmax[r], __shfl_xor(rmax[r], off, 32));
    if (m == 0) {
#pragma unroll
        for (int r = 0; r < 8; ++r) rmx[r + 8 * g] = rmax[r];
    }
    __syncthreads();

    // ---- pass 2: softmax numerators, bucket sums, P @ V via WMMA ----
    float mymax = rmx[m];
    float psum = 0.f;
    v8f ac0 = {}, ac1 = {};
    for (int xt2 = 0; xt2 < Lz / 32; ++xt2) {
        int x0 = xt2 * 32;
        v16h ap;
#pragma unroll
        for (int j = 0; j < 16; ++j) {
            int kk = j + 16 * g;  // key offset in chunk
            float p = __expf(e_lds[m][x0 + kk] - mymax);
            psum += p;
            ap[j] = (f16)p;
            int d = didx[((size_t)b * Lz + l0 + m) * Lz + x0 + kk];
            atomicAdd(&aw[m][d], p);  // ds_add_f32
        }
        v16h bv0, bv1;
#pragma unroll
        for (int j = 0; j < 16; ++j) {
            int key = x0 + j + 16 * g;  // B-fragment K
            const float* vp = v + (((size_t)b * Lz + key) * Hz + h) * KHz;
            bv0[j] = (f16)vp[m];
            bv1[j] = (f16)vp[16 + m];
        }
        ac0 = WMMA(ap, bv0, ac0);
        ac1 = WMMA(ap, bv1, ac1);
    }
    psum += __shfl_xor(psum, 16, 32);
    if (lane < 16) rsum[lane] = psum;
    __syncthreads();

    // ---- epilogue: add bucket value-bias, normalize, store z[B,L,H,KH] ----
#pragma unroll
    for (int r = 0; r < 8; ++r) {
        int row = r + 8 * g;
        float inv = 1.0f / rsum[row];
        float b0v = 0.f, b1v = 0.f;
#pragma unroll
        for (int d = 0; d < NDz; ++d) {
            float w = aw[row][d];
            b0v += w * vb[d][m];
            b1v += w * vb[d][16 + m];
        }
        size_t zb = (((size_t)b * Lz + l0 + row) * Hz + h) * KHz;
        z[zb + m]      = (ac0[r] + b0v) * inv;
        z[zb + 16 + m] = (ac1[r] + b1v) * inv;
    }
}

// ---------- residual + LayerNorm over D=512 ----------
__global__ __launch_bounds__(256) void k_addln(const float* __restrict__ x,
                                               const float* __restrict__ y,
                                               const float* __restrict__ gw,
                                               const float* __restrict__ bw,
                                               float* __restrict__ out) {
    __shared__ float red[256];
    int row = blockIdx.x, t = threadIdx.x;
    const float* xp = x + (size_t)row * Dz;
    const float* yp = y + (size_t)row * Dz;
    float a0 = xp[t] + yp[t];
    float a1 = xp[t + 256] + yp[t + 256];
    red[t] = a0 + a1;
    __syncthreads();
    for (int off = 128; off; off >>= 1) {
        if (t < off) red[t] += red[t + off];
        __syncthreads();
    }
    float mu = red[0] * (1.0f / Dz);
    __syncthreads();
    float d0 = a0 - mu, d1 = a1 - mu;
    red[t] = d0 * d0 + d1 * d1;
    __syncthreads();
    for (int off = 128; off; off >>= 1) {
        if (t < off) red[t] += red[t + off];
        __syncthreads();
    }
    float rs = rsqrtf(red[0] * (1.0f / Dz) + 1e-5f);
    out[(size_t)row * Dz + t]       = d0 * rs * gw[t] + bw[t];
    out[(size_t)row * Dz + t + 256] = d1 * rs * gw[t + 256] + bw[t + 256];
}

__global__ void k_pool(const float* __restrict__ h, float* __restrict__ pooled) {
    int i = blockIdx.x * blockDim.x + threadIdx.x;
    if (i >= Bz * Dz) return;
    int b = i / Dz, d = i % Dz;
    float s = 0.f;
    for (int l = 0; l < Lz; ++l) s += h[((size_t)b * Lz + l) * Dz + d];
    pooled[i] = s;
}

__global__ __launch_bounds__(256) void k_head(const float* __restrict__ pooled,
                                              const float* __restrict__ Wm1,
                                              const float* __restrict__ bm1,
                                              const float* __restrict__ Wm2,
                                              const float* __restrict__ bm2,
                                              float* __restrict__ out) {
    __shared__ float hid[MMz];
    __shared__ float red[256];
    int b = blockIdx.x, t = threadIdx.x;
    const float* pp = pooled + (size_t)b * Dz;
    for (int mm = t; mm < MMz; mm += 256) {
        float s = bm1[mm];
        for (int d = 0; d < Dz; ++d) s += pp[d] * Wm1[(size_t)d * MMz + mm];
        hid[mm] = fmaxf(s, 0.f);
    }
    __syncthreads();
    red[t] = hid[t] * Wm2[t] + hid[t + 256] * Wm2[t + 256];
    __syncthreads();
    for (int off = 128; off; off >>= 1) {
        if (t < off) red[t] += red[t + off];
        __syncthreads();
    }
    if (t == 0) out[b] = red[0] + bm2[0];
}

// ---------- host orchestration ----------
extern "C" void kernel_launch(void* const* d_in, const int* in_sizes, int n_in,
                              void* d_out, int out_size, void* d_ws, size_t ws_size,
                              hipStream_t stream) {
    const int*   ct   = (const int*)d_in[0];
    const float* dist = (const float*)d_in[1];
    const float* emb  = (const float*)d_in[2];
    const float* Kqk  = (const float*)d_in[3];
    const float* Kqr  = (const float*)d_in[4];
    const float* Kkr  = (const float*)d_in[5];
    const float* Vqk  = (const float*)d_in[6];
    const float* Vqr  = (const float*)d_in[7];
    const float* Vkr  = (const float*)d_in[8];
    const float* Wq = (const float*)d_in[9],  *Wk = (const float*)d_in[10];
    const float* Wv = (const float*)d_in[11], *Wo = (const float*)d_in[12];
    const float* bq = (const float*)d_in[13], *bk = (const float*)d_in[14];
    const float* bv = (const float*)d_in[15], *bo = (const float*)d_in[16];
    const float* W1 = (const float*)d_in[17], *b1 = (const float*)d_in[18];
    const float* W2 = (const float*)d_in[19], *b2 = (const float*)d_in[20];
    const float* g1 = (const float*)d_in[21], *be1 = (const float*)d_in[22];
    const float* g2 = (const float*)d_in[23], *be2 = (const float*)d_in[24];
    const float* Wm1 = (const float*)d_in[25], *bm1 = (const float*)d_in[26];
    const float* Wm2 = (const float*)d_in[27], *bm2 = (const float*)d_in[28];

    const size_t SZ_H = (size_t)Bz * Lz * Dz;       // 1 Mi floats
    const size_t SZ_QB = (size_t)Bz * Lz * Hz * NDz;
    float* ws = (float*)d_ws;
    float* h   = ws;
    float* h2  = h + SZ_H;
    float* q   = h2 + SZ_H;
    float* kk  = q + SZ_H;
    float* vv  = kk + SZ_H;
    float* z   = vv + SZ_H;
    float* o   = z + SZ_H;
    float* f1  = o + SZ_H;                           // B*L*FF
    float* f2  = f1 + (size_t)Bz * Lz * FFz;
    float* qb  = f2 + SZ_H;
    float* kb  = qb + SZ_QB;
    float* pooled = kb + SZ_QB;
    unsigned char* didx = (unsigned char*)(pooled + (size_t)Bz * Dz);

    const int BL = Bz * Lz;  // 2048

    k_embed<<<(Bz * Lz * Dz + 255) / 256, 256, 0, stream>>>(ct, emb, h);
    k_didx<<<(Bz * Lz * Lz + 255) / 256, 256, 0, stream>>>(dist, didx);

    dim3 gqkv(Dz / 64, BL / 64);     // (8, 32)
    dim3 gff1(FFz / 64, BL / 64);    // (32, 32)
    dim3 gff2(Dz / 64, BL / 64);     // (8, 32)
    int relN = Bz * Lz * Hz * NDz;

    for (int li = 0; li < NL; ++li) {
        const float* wq = Wq + (size_t)li * Dz * Dz;
        const float* wk = Wk + (size_t)li * Dz * Dz;
        const float* wv = Wv + (size_t)li * Dz * Dz;
        const float* wo = Wo + (size_t)li * Dz * Dz;
        const float* w1 = W1 + (size_t)li * Dz * FFz;
        const float* w2 = W2 + (size_t)li * FFz * Dz;

        k_gemm<0><<<gqkv, 256, 0, stream>>>(h, wq, bq + li * Dz, q, BL, Dz, Dz);
        k_gemm<0><<<gqkv, 256, 0, stream>>>(h, wk, bk + li * Dz, kk, BL, Dz, Dz);
        k_gemm<0><<<gqkv, 256, 0, stream>>>(h, wv, bv + li * Dz, vv, BL, Dz, Dz);

        k_relproj<<<(relN + 255) / 256, 256, 0, stream>>>(q, Kqk, qb);
        k_relproj<<<(relN + 255) / 256, 256, 0, stream>>>(kk, Kqr, kb);

        k_attn<<<Bz * Hz * (Lz / 16), 32, 0, stream>>>(q, kk, vv, qb, kb, didx,
                                                       Kkr, Vqk, Vqr, Vkr, z);

        k_gemm<0><<<gqkv, 256, 0, stream>>>(z, wo, bo + li * Dz, o, BL, Dz, Dz);
        k_addln<<<BL, 256, 0, stream>>>(h, o, g1 + li * Dz, be1 + li * Dz, h2);

        k_gemm<1><<<gff1, 256, 0, stream>>>(h2, w1, b1 + li * FFz, f1, BL, FFz, Dz);
        k_gemm<0><<<gff2, 256, 0, stream>>>(f1, w2, b2 + li * Dz, f2, BL, Dz, FFz);
        k_addln<<<BL, 256, 0, stream>>>(h2, f2, g2 + li * Dz, be2 + li * Dz, h);
    }

    k_pool<<<(Bz * Dz + 255) / 256, 256, 0, stream>>>(h, pooled);
    k_head<<<Bz, 256, 0, stream>>>(pooled, Wm1, bm1, Wm2, bm2, (float*)d_out);
}
